// GCLConv_75024488726858
// MI455X (gfx1250) — matrix operand order repaired and checked by reading, not compiled, not run on hardware
//
#include <hip/hip_runtime.h>
#include <hip/hip_bf16.h>

// ---------- types for WMMA fragments ----------
typedef __attribute__((ext_vector_type(16))) __bf16 bf16x16;
typedef __attribute__((ext_vector_type(8)))  float  floatx8;

union BF16Frag {
    bf16x16        v;
    uint4          q[2];
    unsigned short s[16];
};

__device__ __forceinline__ unsigned short f32_to_bf16_rne(float f) {
    unsigned u = __float_as_uint(f);
    u += 0x7FFFu + ((u >> 16) & 1u);
    return (unsigned short)(u >> 16);
}
__device__ __forceinline__ float siluf(float x)  { return x / (1.f + __expf(-x)); }
__device__ __forceinline__ float sigmf(float x)  { return 1.f / (1.f + __expf(-x)); }

#define M1_STRIDE 136   // 16B-padded row stride (ushort units) -> conflict-free b128 LDS loads
#define M2_STRIDE 132   // padded f32 row stride -> conflict-free b32 LDS access
#define EDGE_WAVES 2    // waves per block in edge kernel (LDS budget)
#define NODE_WAVES 4    // waves per block in node kernel

// ---------- prep kernels ----------
__global__ void zero_f32_kernel(float* __restrict__ p, int n) {
    int i = blockIdx.x * blockDim.x + threadIdx.x;
    if (i < n) p[i] = 0.f;
}

__global__ void cvt_bf16_kernel(const float* __restrict__ src,
                                unsigned short* __restrict__ dst, int n) {
    int i = blockIdx.x * blockDim.x + threadIdx.x;
    if (i < n) dst[i] = f32_to_bf16_rne(src[i]);
}

// Pack W[fanin][128] (row-major) into WMMA-B fragment order:
// block (nt,kt) holds a 32x16 (KxN) slab; lane l keeps column n = nt*16 + (l&15),
// elements j=0..15 -> k = kt*32 + (j>>3)*16 + (l>>4)*8 + (j&7), stored contiguously (32 B / lane).
__global__ void pack_weight_kernel(const float* __restrict__ W,
                                   unsigned short* __restrict__ out, int fanin) {
    int t = blockIdx.x * blockDim.x + threadIdx.x;
    int total = fanin * 128;
    if (t >= total) return;
    int j    = t & 15;
    int lane = (t >> 4) & 31;
    int rest = t >> 9;              // nt*KT + kt
    int KT   = fanin >> 5;
    int kt   = rest % KT;
    int nt   = rest / KT;
    int n = nt * 16 + (lane & 15);
    int k = kt * 32 + (j >> 3) * 16 + (lane >> 4) * 8 + (j & 7);
    out[t] = f32_to_bf16_rne(W[k * 128 + n]);
}

// ---------- edge kernel: per-wave 32-edge tile (B fragment shared by 2 WMMAs) ----------
__global__ __launch_bounds__(32 * EDGE_WAVES)
void edge_kernel(const unsigned short* __restrict__ h_bf,
                 const long long*      __restrict__ eidx,
                 const unsigned short* __restrict__ eW1p,
                 const float*          __restrict__ eb1,
                 const unsigned short* __restrict__ eW2p,
                 const float*          __restrict__ eb2,
                 const float*          __restrict__ aW,
                 const float*          __restrict__ abp,
                 float*                __restrict__ agg,
                 int E, int nTiles)
{
    __shared__ __align__(16) unsigned short lds_m1[EDGE_WAVES][32 * M1_STRIDE]; // 17.4 KB
    __shared__ __align__(16) float          lds_m2[EDGE_WAVES][32 * M2_STRIDE]; // 33.8 KB

    const int lane = threadIdx.x & 31;
    const int wave = threadIdx.x >> 5;
    const int tile = blockIdx.x * EDGE_WAVES + wave;
    if (tile >= nTiles) return;

    const int nlow = lane & 15;
    const int hi   = lane >> 4;

    // the two edges whose feature rows this lane gathers for A fragments
    const long long ebase = (long long)tile * 32;
    const long long rn0 = eidx[ebase + nlow];
    const long long cn0 = eidx[(long long)E + ebase + nlow];
    const long long rn1 = eidx[ebase + 16 + nlow];
    const long long cn1 = eidx[(long long)E + ebase + 16 + nlow];

    unsigned short* m1 = lds_m1[wave];
    float*          m2 = lds_m2[wave];

    // ---- layer 1: x(32x256) @ eW1(256x128) + eb1, SiLU -> m1 in LDS ----
    for (int nt = 0; nt < 8; ++nt) {
        float bias = eb1[nt * 16 + nlow];
        floatx8 c0, c1;
        #pragma unroll
        for (int r = 0; r < 8; ++r) { c0[r] = bias; c1[r] = bias; }

        #pragma unroll
        for (int kt = 0; kt < 8; ++kt) {
            const long long n0 = (kt < 4) ? rn0 : cn0;
            const long long n1 = (kt < 4) ? rn1 : cn1;
            const int ksub = (kt & 3) * 32 + hi * 8;
            const unsigned short* ap0 = h_bf + n0 * 128 + ksub;
            const unsigned short* ap1 = h_bf + n1 * 128 + ksub;
            BF16Frag a0, a1;
            a0.q[0] = *(const uint4*)(ap0);
            a0.q[1] = *(const uint4*)(ap0 + 16);
            a1.q[0] = *(const uint4*)(ap1);
            a1.q[1] = *(const uint4*)(ap1 + 16);
            const unsigned short* bp = eW1p + (((nt * 8 + kt) * 32 + lane) << 4);
            BF16Frag b;
            b.q[0] = *(const uint4*)(bp);
            b.q[1] = *(const uint4*)(bp + 8);
            c0 = __builtin_amdgcn_wmma_f32_16x16x32_bf16(false, a0.v, false, b.v,
                                                         (short)0, c0, false, false);
            c1 = __builtin_amdgcn_wmma_f32_16x16x32_bf16(false, a1.v, false, b.v,
                                                         (short)0, c1, false, false);
        }
        #pragma unroll
        for (int r = 0; r < 8; ++r) {
            m1[(r + 8 * hi) * M1_STRIDE + nt * 16 + nlow]        = f32_to_bf16_rne(siluf(c0[r]));
            m1[(16 + r + 8 * hi) * M1_STRIDE + nt * 16 + nlow]   = f32_to_bf16_rne(siluf(c1[r]));
        }
    }
    __builtin_amdgcn_wave_barrier();   // LDS is wave-private; same-wave DS ops are in-order

    // ---- layer 2: m1(32x128) @ eW2(128x128) + eb2, SiLU -> m2 LDS; attention partials ----
    float attacc0[8], attacc1[8];
    #pragma unroll
    for (int r = 0; r < 8; ++r) { attacc0[r] = 0.f; attacc1[r] = 0.f; }

    for (int nt = 0; nt < 8; ++nt) {
        float bias = eb2[nt * 16 + nlow];
        floatx8 c0, c1;
        #pragma unroll
        for (int r = 0; r < 8; ++r) { c0[r] = bias; c1[r] = bias; }

        #pragma unroll
        for (int kt = 0; kt < 4; ++kt) {
            const unsigned short* ap0 = m1 + nlow * M1_STRIDE + kt * 32 + hi * 8;
            const unsigned short* ap1 = m1 + (16 + nlow) * M1_STRIDE + kt * 32 + hi * 8;
            BF16Frag a0, a1;
            a0.q[0] = *(const uint4*)(ap0);
            a0.q[1] = *(const uint4*)(ap0 + 16);
            a1.q[0] = *(const uint4*)(ap1);
            a1.q[1] = *(const uint4*)(ap1 + 16);
            const unsigned short* bp = eW2p + (((nt * 4 + kt) * 32 + lane) << 4);
            BF16Frag b;
            b.q[0] = *(const uint4*)(bp);
            b.q[1] = *(const uint4*)(bp + 8);
            c0 = __builtin_amdgcn_wmma_f32_16x16x32_bf16(false, a0.v, false, b.v,
                                                         (short)0, c0, false, false);
            c1 = __builtin_amdgcn_wmma_f32_16x16x32_bf16(false, a1.v, false, b.v,
                                                         (short)0, c1, false, false);
        }
        float wn = aW[nt * 16 + nlow];
        #pragma unroll
        for (int r = 0; r < 8; ++r) {
            float v0 = siluf(c0[r]);
            float v1 = siluf(c1[r]);
            m2[(r + 8 * hi) * M2_STRIDE + nt * 16 + nlow]      = v0;
            m2[(16 + r + 8 * hi) * M2_STRIDE + nt * 16 + nlow] = v1;
            attacc0[r] += v0 * wn;
            attacc1[r] += v1 * wn;
        }
    }

    // ---- att = sigmoid(m2 @ aW + ab): reduce across the 16 lanes sharing a row ----
    const float ab0 = abp[0];
    float att0[8], att1[8];
    #pragma unroll
    for (int r = 0; r < 8; ++r) {
        float s0 = attacc0[r];
        s0 += __shfl_xor(s0, 1, 32);
        s0 += __shfl_xor(s0, 2, 32);
        s0 += __shfl_xor(s0, 4, 32);
        s0 += __shfl_xor(s0, 8, 32);
        att0[r] = sigmf(s0 + ab0) * 0.01f;   // fold 1/NORM_FACTOR into the gate
        float s1 = attacc1[r];
        s1 += __shfl_xor(s1, 1, 32);
        s1 += __shfl_xor(s1, 2, 32);
        s1 += __shfl_xor(s1, 4, 32);
        s1 += __shfl_xor(s1, 8, 32);
        att1[r] = sigmf(s1 + ab0) * 0.01f;
    }
    __builtin_amdgcn_wave_barrier();

    // ---- scatter: agg[row[e]] += m2 * att (read-back pattern == write pattern) ----
    #pragma unroll
    for (int r = 0; r < 8; ++r) {
        {
            const int m = r + 8 * hi;
            const long long rowm = eidx[ebase + m];
            float* dst = agg + rowm * 128 + nlow;
            const float av = att0[r];
            #pragma unroll
            for (int nt = 0; nt < 8; ++nt)
                unsafeAtomicAdd(dst + nt * 16, m2[m * M2_STRIDE + nt * 16 + nlow] * av);
        }
        {
            const int m = 16 + r + 8 * hi;
            const long long rowm = eidx[ebase + m];
            float* dst = agg + rowm * 128 + nlow;
            const float av = att1[r];
            #pragma unroll
            for (int nt = 0; nt < 8; ++nt)
                unsafeAtomicAdd(dst + nt * 16, m2[m * M2_STRIDE + nt * 16 + nlow] * av);
        }
    }
}

// ---------- node kernel: per-wave 16-node tile ----------
__global__ __launch_bounds__(32 * NODE_WAVES)
void node_kernel(const float*          __restrict__ h,
                 const unsigned short* __restrict__ h_bf,
                 const float*          __restrict__ agg,
                 const unsigned short* __restrict__ nW1p,
                 const float*          __restrict__ nb1,
                 const unsigned short* __restrict__ nW2p,
                 const float*          __restrict__ nb2,
                 float*                __restrict__ out,
                 int nTiles)
{
    __shared__ __align__(16) unsigned short lds_z1[NODE_WAVES][16 * M1_STRIDE];

    const int lane = threadIdx.x & 31;
    const int wave = threadIdx.x >> 5;
    const int tile = blockIdx.x * NODE_WAVES + wave;
    if (tile >= nTiles) return;

    const int nlow = lane & 15;
    const int hi   = lane >> 4;
    const long long node = (long long)tile * 16 + nlow;

    unsigned short* z1 = lds_z1[wave];

    // ---- layer 1: [h | agg](16x256) @ nW1 + nb1, SiLU -> z1 in LDS ----
    for (int nt = 0; nt < 8; ++nt) {
        float bias = nb1[nt * 16 + nlow];
        floatx8 c;
        #pragma unroll
        for (int r = 0; r < 8; ++r) c[r] = bias;

        #pragma unroll
        for (int kt = 0; kt < 8; ++kt) {
            BF16Frag a;
            if (kt < 4) {
                const unsigned short* ap = h_bf + node * 128 + kt * 32 + hi * 8;
                a.q[0] = *(const uint4*)(ap);
                a.q[1] = *(const uint4*)(ap + 16);
            } else {
                const float* gp = agg + node * 128 + (kt - 4) * 32 + hi * 8;
                float4 f0 = *(const float4*)(gp);
                float4 f1 = *(const float4*)(gp + 4);
                float4 f2 = *(const float4*)(gp + 16);
                float4 f3 = *(const float4*)(gp + 20);
                a.s[0]  = f32_to_bf16_rne(f0.x); a.s[1]  = f32_to_bf16_rne(f0.y);
                a.s[2]  = f32_to_bf16_rne(f0.z); a.s[3]  = f32_to_bf16_rne(f0.w);
                a.s[4]  = f32_to_bf16_rne(f1.x); a.s[5]  = f32_to_bf16_rne(f1.y);
                a.s[6]  = f32_to_bf16_rne(f1.z); a.s[7]  = f32_to_bf16_rne(f1.w);
                a.s[8]  = f32_to_bf16_rne(f2.x); a.s[9]  = f32_to_bf16_rne(f2.y);
                a.s[10] = f32_to_bf16_rne(f2.z); a.s[11] = f32_to_bf16_rne(f2.w);
                a.s[12] = f32_to_bf16_rne(f3.x); a.s[13] = f32_to_bf16_rne(f3.y);
                a.s[14] = f32_to_bf16_rne(f3.z); a.s[15] = f32_to_bf16_rne(f3.w);
            }
            const unsigned short* bp = nW1p + (((nt * 8 + kt) * 32 + lane) << 4);
            BF16Frag b;
            b.q[0] = *(const uint4*)(bp);
            b.q[1] = *(const uint4*)(bp + 8);
            c = __builtin_amdgcn_wmma_f32_16x16x32_bf16(false, a.v, false, b.v,
                                                        (short)0, c, false, false);
        }
        #pragma unroll
        for (int r = 0; r < 8; ++r) {
            z1[(r + 8 * hi) * M1_STRIDE + nt * 16 + nlow] = f32_to_bf16_rne(siluf(c[r]));
        }
    }
    __builtin_amdgcn_wave_barrier();

    // ---- layer 2: z1(16x128) @ nW2(128x128) + nb2 + residual ----
    for (int nt = 0; nt < 8; ++nt) {
        float bias = nb2[nt * 16 + nlow];
        floatx8 c;
        #pragma unroll
        for (int r = 0; r < 8; ++r) c[r] = bias;

        #pragma unroll
        for (int kt = 0; kt < 4; ++kt) {
            const unsigned short* ap = z1 + nlow * M1_STRIDE + kt * 32 + hi * 8;
            BF16Frag a;
            a.q[0] = *(const uint4*)(ap);
            a.q[1] = *(const uint4*)(ap + 16);
            const unsigned short* bp = nW2p + (((nt * 4 + kt) * 32 + lane) << 4);
            BF16Frag b;
            b.q[0] = *(const uint4*)(bp);
            b.q[1] = *(const uint4*)(bp + 8);
            c = __builtin_amdgcn_wmma_f32_16x16x32_bf16(false, a.v, false, b.v,
                                                        (short)0, c, false, false);
        }
        #pragma unroll
        for (int r = 0; r < 8; ++r) {
            const long long n2 = (long long)tile * 16 + r + 8 * hi;
            const long long idx = n2 * 128 + nt * 16 + nlow;
            out[idx] = h[idx] + c[r];
        }
    }
}

// ---------- host launcher ----------
extern "C" void kernel_launch(void* const* d_in, const int* in_sizes, int n_in,
                              void* d_out, int out_size, void* d_ws, size_t ws_size,
                              hipStream_t stream) {
    const float*     h    = (const float*)d_in[0];
    const long long* eidx = (const long long*)d_in[1];   // int64 [2,E]
    const float*     eW1  = (const float*)d_in[2];
    const float*     eb1  = (const float*)d_in[3];
    const float*     eW2  = (const float*)d_in[4];
    const float*     eb2  = (const float*)d_in[5];
    const float*     aW   = (const float*)d_in[6];
    const float*     abp  = (const float*)d_in[7];
    const float*     nW1  = (const float*)d_in[8];
    const float*     nb1  = (const float*)d_in[9];
    const float*     nW2  = (const float*)d_in[10];
    const float*     nb2  = (const float*)d_in[11];
    float* out = (float*)d_out;

    const int N = in_sizes[0] / 128;   // 50000
    const int E = in_sizes[1] / 2;     // 800000
    const int NH = N * 128;

    // workspace carve-up (256-B aligned pieces)
    char* ws = (char*)d_ws;
    size_t off = 0;
    auto carve = [&](size_t bytes) {
        void* p = ws + off;
        off += (bytes + 255) & ~(size_t)255;
        return p;
    };
    unsigned short* h_bf = (unsigned short*)carve((size_t)NH * 2);
    float*          agg  = (float*)carve((size_t)NH * 4);
    unsigned short* eW1p = (unsigned short*)carve(256 * 128 * 2);
    unsigned short* eW2p = (unsigned short*)carve(128 * 128 * 2);
    unsigned short* nW1p = (unsigned short*)carve(256 * 128 * 2);
    unsigned short* nW2p = (unsigned short*)carve(128 * 128 * 2);

    // prep: zero accumulator, bf16 copies, weight packing
    zero_f32_kernel<<<(NH + 255) / 256, 256, 0, stream>>>(agg, NH);
    cvt_bf16_kernel<<<(NH + 255) / 256, 256, 0, stream>>>(h, h_bf, NH);
    pack_weight_kernel<<<(256 * 128) / 256, 256, 0, stream>>>(eW1, eW1p, 256);
    pack_weight_kernel<<<(128 * 128) / 256, 256, 0, stream>>>(eW2, eW2p, 128);
    pack_weight_kernel<<<(256 * 128) / 256, 256, 0, stream>>>(nW1, nW1p, 256);
    pack_weight_kernel<<<(128 * 128) / 256, 256, 0, stream>>>(nW2, nW2p, 128);

    // edge phase: one 32-edge tile per wave
    const int eTiles = (E + 31) / 32;                       // 25000
    const int eBlocks = (eTiles + EDGE_WAVES - 1) / EDGE_WAVES;
    edge_kernel<<<eBlocks, 32 * EDGE_WAVES, 0, stream>>>(
        h_bf, eidx, eW1p, eb1, eW2p, eb2, aW, abp, agg, E, eTiles);

    // node phase: one 16-node tile per wave
    const int nTiles = (N + 15) / 16;                       // 3125
    const int nBlocks = (nTiles + NODE_WAVES - 1) / NODE_WAVES;
    node_kernel<<<nBlocks, 32 * NODE_WAVES, 0, stream>>>(
        h, h_bf, agg, nW1p, nb1, nW2p, nb2, out, nTiles);
}
